// MessageFunction_46892452938051
// MI455X (gfx1250) — matrix operand order repaired and checked by reading, not compiled, not run on hardware
//
#include <hip/hip_runtime.h>

// ---------------------------------------------------------------------------
// msg[b,o,n] = sum_d We[o,d]*e_vw[b,d,n] + sum_d Ww[o,d]*h_w[b,d,n] + be[o]+bw[o]
// Treated as C(768x256) = Wc(768x1536) x Xb(1536x256) per batch, bf16 WMMA,
// fp32 accumulate.  Memory-bound target: ~307 MB of HBM traffic @ 23.3 TB/s.
// ---------------------------------------------------------------------------

typedef __attribute__((ext_vector_type(16))) __bf16 v16bf;
typedef __attribute__((ext_vector_type(8)))  __bf16 v8bf;
typedef __attribute__((ext_vector_type(8)))  float  v8f;

#define D_DIM 768
#define N_DIM 256
#define K_DIM 1536          // 2*D: [We | Ww] along k
#define TM 128
#define TN 128
#define TK 32
#define LDSS 40             // padded LDS row stride in halves (80B -> conflict-free-ish)
#define NKSTEPS (K_DIM / TK)  // 48

// round-to-nearest-even fp32 -> bf16, packed pair into one dword
__device__ __forceinline__ unsigned bfpair(float a, float b) {
    union U { float f; unsigned u; };
    U x; x.f = a;
    U y; y.f = b;
    unsigned lo = (x.u + 0x7FFFu + ((x.u >> 16) & 1u)) >> 16;
    unsigned hi = (y.u + 0x7FFFu + ((y.u >> 16) & 1u)) & 0xFFFF0000u;
    return hi | lo;
}

// Stage one K-step: A tile (128 rows x 32 k, weights, row-major k-contig) and
// B tile (32 k x 128 n activations, transposed into Bs[n][k]), both fp32->bf16.
__device__ __forceinline__ void stage_tiles(
    unsigned short* __restrict__ As, unsigned short* __restrict__ Bs,
    const float* __restrict__ Wsrc,   // points at [o=0][k-slice base] of the weight matrix
    const float* __restrict__ Xsrc,   // points at [k-slice base][n=0] of the activation matrix
    int mT, int nT, int t)
{
    // ---- A: thread t covers row t>>1, k-half (t&1)*16 ----
    {
        const int row = t >> 1;
        const int kh  = (t & 1) * 16;
        const float* s = Wsrc + (size_t)(mT + row) * D_DIM + kh;
        float4 f0 = *(const float4*)(s + 0);
        float4 f1 = *(const float4*)(s + 4);
        float4 f2 = *(const float4*)(s + 8);
        float4 f3 = *(const float4*)(s + 12);
        uint4 u0, u1;
        u0.x = bfpair(f0.x, f0.y); u0.y = bfpair(f0.z, f0.w);
        u0.z = bfpair(f1.x, f1.y); u0.w = bfpair(f1.z, f1.w);
        u1.x = bfpair(f2.x, f2.y); u1.y = bfpair(f2.z, f2.w);
        u1.z = bfpair(f3.x, f3.y); u1.w = bfpair(f3.z, f3.w);
        *(uint4*)&As[row * LDSS + kh]     = u0;   // 16B aligned (stride 80B)
        *(uint4*)&As[row * LDSS + kh + 8] = u1;
    }
    // ---- B: 512 units of (k-pair, 4 n); 2 units per thread.  Transposed store:
    //      consecutive k land adjacent in Bs[n][k] -> packed b32 LDS stores. ----
#pragma unroll
    for (int u = 0; u < 2; ++u) {
        const int id = t * 2 + u;
        const int kp = (id & 15) * 2;      // k base (even), 0..30
        const int ng = (id >> 4) * 4;      // n base, 0..124
        const float* s0 = Xsrc + (size_t)kp * N_DIM + nT + ng;
        float4 a = *(const float4*)s0;
        float4 c = *(const float4*)(s0 + N_DIM);
        *(unsigned*)&Bs[(ng + 0) * LDSS + kp] = bfpair(a.x, c.x);
        *(unsigned*)&Bs[(ng + 1) * LDSS + kp] = bfpair(a.y, c.y);
        *(unsigned*)&Bs[(ng + 2) * LDSS + kp] = bfpair(a.z, c.z);
        *(unsigned*)&Bs[(ng + 3) * LDSS + kp] = bfpair(a.w, c.w);
    }
}

__global__ __launch_bounds__(256)
void msg_gemm_kernel(const float* __restrict__ h_w,
                     const float* __restrict__ e_vw,
                     const float* __restrict__ We,
                     const float* __restrict__ be,
                     const float* __restrict__ Ww,
                     const float* __restrict__ bw,
                     float* __restrict__ out)
{
    __shared__ __align__(16) unsigned short As[2][TM * LDSS];  // 10 KB each
    __shared__ __align__(16) unsigned short Bs[2][TN * LDSS];  // 40 KB total

    const int b    = blockIdx.y;
    const int mT   = (blockIdx.x % 6) * TM;   // 6 M-tiles of 128 over 768
    const int nT   = (blockIdx.x / 6) * TN;   // 2 N-tiles of 128 over 256
    const int t    = threadIdx.x;
    const int wave = t >> 5;
    const int lane = t & 31;
    const int wM   = (wave >> 1) * 32;        // wave grid 4(M) x 2(N)
    const int wN   = (wave & 1) * 64;

    const float* Xe = e_vw + (size_t)b * D_DIM * N_DIM;
    const float* Xh = h_w  + (size_t)b * D_DIM * N_DIM;

    v8f acc[2][4];
#pragma unroll
    for (int mi = 0; mi < 2; ++mi)
#pragma unroll
        for (int ni = 0; ni < 4; ++ni)
            acc[mi][ni] = (v8f){0.f, 0.f, 0.f, 0.f, 0.f, 0.f, 0.f, 0.f};

    // Prologue: stage K-step 0 into buffer 0 (k0=0 -> We / e_vw half).
    stage_tiles(As[0], Bs[0], We, Xe, mT, nT, t);

    // Per-lane fragment addressing (ISA 7.12.2 layouts, wave32):
    //  A 16x32 bf16: lanes<16 hold K 0-7 & 16-23, lanes>=16 hold K 8-15 & 24-31
    //  B 32x16 bf16: lanes<16 hold K 0-7 & 16-23 of col n, lanes>=16 hold K 8-15 & 24-31
    const int lrow = lane & 15;
    const int aKh  = (lane >> 4) * 8;
    const int bKl  = (lane >> 4) * 8;

    for (int kt = 0; kt < NKSTEPS; ++kt) {
        __syncthreads();   // prev stage visible; prev compute done -> safe overwrite

        if (kt + 1 < NKSTEPS) {
            const int  k0    = (kt + 1) * TK;
            const bool first = (k0 < D_DIM);            // TK divides 768: no straddle
            const float* Wsrc = first ? (We + k0) : (Ww + (k0 - D_DIM));
            const float* Xsrc = first ? (Xe + (size_t)k0 * N_DIM)
                                      : (Xh + (size_t)(k0 - D_DIM) * N_DIM);
            stage_tiles(As[(kt + 1) & 1], Bs[(kt + 1) & 1], Wsrc, Xsrc, mT, nT, t);
        }

        const unsigned short* Ac = As[kt & 1];
        const unsigned short* Bc = Bs[kt & 1];

        v16bf aF[2], bF[4];
#pragma unroll
        for (int mi = 0; mi < 2; ++mi) {
            const unsigned short* p = &Ac[(wM + mi * 16 + lrow) * LDSS + aKh];
            v8bf lo = *(const v8bf*)(p);        // K k..k+7
            v8bf hi = *(const v8bf*)(p + 16);   // K k+16..k+23
            aF[mi] = __builtin_shufflevector(lo, hi,
                     0, 1, 2, 3, 4, 5, 6, 7, 8, 9, 10, 11, 12, 13, 14, 15);
        }
#pragma unroll
        for (int ni = 0; ni < 4; ++ni) {
            const unsigned short* p = &Bc[(wN + ni * 16 + lrow) * LDSS + bKl];
            v8bf lo = *(const v8bf*)(p);        // K contiguous (transposed tile)
            v8bf hi = *(const v8bf*)(p + 16);
            bF[ni] = __builtin_shufflevector(lo, hi,
                     0, 1, 2, 3, 4, 5, 6, 7, 8, 9, 10, 11, 12, 13, 14, 15);
        }

#pragma unroll
        for (int mi = 0; mi < 2; ++mi)
#pragma unroll
            for (int ni = 0; ni < 4; ++ni)
                acc[mi][ni] = __builtin_amdgcn_wmma_f32_16x16x32_bf16(
                    /*neg_a=*/false, aF[mi], /*neg_b=*/false, bF[ni],
                    /*c_mod=*/(short)0, acc[mi][ni],
                    /*reuse_a=*/false, /*reuse_b=*/false);
    }

    // Epilogue: C layout -> VGPR j is row (j | +8 for upper lane half), col = lane&15.
    const int mhalf = (lane >> 4) * 8;
#pragma unroll
    for (int mi = 0; mi < 2; ++mi) {
#pragma unroll
        for (int j = 0; j < 8; ++j) {
            const int   r    = mT + wM + mi * 16 + mhalf + j;
            const float bias = be[r] + bw[r];
            float* orow = out + ((size_t)b * D_DIM + r) * N_DIM;
#pragma unroll
            for (int ni = 0; ni < 4; ++ni)
                orow[nT + wN + ni * 16 + lrow] = acc[mi][ni][j] + bias;
        }
    }
}

extern "C" void kernel_launch(void* const* d_in, const int* in_sizes, int n_in,
                              void* d_out, int out_size, void* d_ws, size_t ws_size,
                              hipStream_t stream) {
    // setup_inputs order: 0=h_v (unused), 1=h_w, 2=e_vw, 3=We, 4=be, 5=Ww, 6=bw
    const float* h_w  = (const float*)d_in[1];
    const float* e_vw = (const float*)d_in[2];
    const float* We   = (const float*)d_in[3];
    const float* be   = (const float*)d_in[4];
    const float* Ww   = (const float*)d_in[5];
    const float* bw   = (const float*)d_in[6];
    float* out = (float*)d_out;

    const int B = in_sizes[1] / (D_DIM * N_DIM);   // 128
    dim3 grid(6 * (N_DIM / TN), B);                // 12 tiles x 128 batches
    msg_gemm_kernel<<<grid, 256, 0, stream>>>(h_w, e_vw, We, be, Ww, bw, out);
}
